// SparseAttention_55018531062340
// MI455X (gfx1250) — compile-verified, matrix operand order checked by reading, not run on hardware
//
#include <hip/hip_runtime.h>
#include <hip/hip_bf16.h>

typedef __attribute__((ext_vector_type(16))) __bf16 v16bf;
typedef __attribute__((ext_vector_type(8)))  __bf16 v8bf;
typedef __attribute__((ext_vector_type(8)))  float  v8f;
typedef __attribute__((ext_vector_type(4)))  float  v4f;

#define S_LEN 4096
#define EMB   1024
#define HEADS 16
#define HD    64
#define BLK   64
#define MAXE  16

// ---------------------------------------------------------------------------
// WMMA helper (CDNA5 wave32, V_WMMA_F32_16X16X32_BF16)
// ---------------------------------------------------------------------------
static __device__ inline v8f wmma_bf16(v16bf a, v16bf b, v8f c) {
  return __builtin_amdgcn_wmma_f32_16x16x32_bf16(
      /*neg_a=*/false, a, /*neg_b=*/false, b,
      /*c_mod=*/(short)0, c, /*reuse_a=*/false, /*reuse_b=*/false);
}

// A/B 16-bit tile layout (ISA 7.12.2): lane L holds row (L&15); quad q of
// VGPRs holds K = q*16 + (L>>4)*8 + 0..7  => two 16-byte contiguous loads.
static __device__ inline v16bf tile_load_bf16(const __bf16* base, int stride,
                                              int row0, int col0, int lane) {
  const int r = row0 + (lane & 15);
  const int c = col0 + ((lane >> 4) << 3);
  const __bf16* p = base + (size_t)r * stride + c;
  v8bf lo = *(const v8bf*)p;
  v8bf hi = *(const v8bf*)(p + 16);
  v16bf o;
#pragma unroll
  for (int i = 0; i < 8; ++i) { o[i] = lo[i]; o[i + 8] = hi[i]; }
  return o;
}

// ---------------------------------------------------------------------------
// One-time fp32 -> bf16 conversion (x and the four weight matrices).
// ---------------------------------------------------------------------------
__global__ void cvt_bf16_kernel(const float* __restrict__ src,
                                __bf16* __restrict__ dst, long n8) {
  long i = (long)blockIdx.x * blockDim.x + threadIdx.x;
  const long stride = (long)gridDim.x * blockDim.x;
  for (; i < n8; i += stride) {
    v4f a = *(const v4f*)(src + i * 8);
    v4f b = *(const v4f*)(src + i * 8 + 4);
    v8bf o;
#pragma unroll
    for (int j = 0; j < 4; ++j) { o[j] = (__bf16)a[j]; o[j + 4] = (__bf16)b[j]; }
    *(v8bf*)(dst + i * 8) = o;
  }
}

// ---------------------------------------------------------------------------
// Zero-fill attn output (exactly 0 outside the sparse pattern). 1 GB stream:
// non-temporal stores so we don't evict the L2-resident bf16 working set.
// ---------------------------------------------------------------------------
__global__ void zero_f32_kernel(v4f* __restrict__ p, long n4) {
  long i = (long)blockIdx.x * blockDim.x + threadIdx.x;
  const long stride = (long)gridDim.x * blockDim.x;
  v4f z = {};
  for (; i < n4; i += stride) __builtin_nontemporal_store(z, &p[i]);
}

// ---------------------------------------------------------------------------
// Projection GEMM (bf16 A and W, fp32 accumulate):
//   Y[s,f] = (sum_e A[s,e] * W[f,e] + b[f]) * out_scale
// 128 threads (4 waves); each wave owns a 32x64 output tile (2 M-subtiles x
// 4 N-subtiles = 8 WMMA per 32-deep K chunk) => workgroup covers 128x64.
// Output: bf16 [S][E], bf16 transposed [E][S] (for V), or fp32 (final proj).
// ---------------------------------------------------------------------------
__global__ __launch_bounds__(128) void proj_gemm_kernel(
    const __bf16* __restrict__ A, const __bf16* __restrict__ W,
    const float* __restrict__ bias,
    __bf16* __restrict__ out_bf, float* __restrict__ out_f32,
    float out_scale, int transpose_out)
{
  const int lane = threadIdx.x & 31;
  const int wave = threadIdx.x >> 5;
  const int tx = blockIdx.x & 15;   // N tile (16 tiles of 64 cols)
  const int ty = blockIdx.x >> 4;   // M tile (32 tiles of 128 rows)
  const int row0 = ty * 128 + wave * 32;
  const int n0 = tx * 64;

  v8f acc[2][4] = {};
  for (int kc = 0; kc < EMB / 32; ++kc) {
    const int kb = kc * 32;
    // Issue every load of the chunk up front; WMMAs drain them with
    // partial loadcnt waits while later loads are still in flight.
    v16bf a0 = tile_load_bf16(A, EMB, row0, kb, lane);
    v16bf a1 = tile_load_bf16(A, EMB, row0 + 16, kb, lane);
    v16bf b0 = tile_load_bf16(W, EMB, n0, kb, lane);
    v16bf b1 = tile_load_bf16(W, EMB, n0 + 16, kb, lane);
    v16bf b2 = tile_load_bf16(W, EMB, n0 + 32, kb, lane);
    v16bf b3 = tile_load_bf16(W, EMB, n0 + 48, kb, lane);
    __builtin_prefetch(A + (size_t)(row0 + (lane & 15)) * EMB + kb + 32, 0, 1);
    acc[0][0] = wmma_bf16(a0, b0, acc[0][0]);
    acc[0][1] = wmma_bf16(a0, b1, acc[0][1]);
    acc[0][2] = wmma_bf16(a0, b2, acc[0][2]);
    acc[0][3] = wmma_bf16(a0, b3, acc[0][3]);
    acc[1][0] = wmma_bf16(a1, b0, acc[1][0]);
    acc[1][1] = wmma_bf16(a1, b1, acc[1][1]);
    acc[1][2] = wmma_bf16(a1, b2, acc[1][2]);
    acc[1][3] = wmma_bf16(a1, b3, acc[1][3]);
  }

  const int n_lane = lane & 15;
  const int m_hi = (lane >> 4) << 3;
#pragma unroll
  for (int mt = 0; mt < 2; ++mt) {
#pragma unroll
    for (int nt = 0; nt < 4; ++nt) {
      const int col = n0 + nt * 16 + n_lane;
      const float bcol = bias[col];
#pragma unroll
      for (int v = 0; v < 8; ++v) {
        const int row = row0 + mt * 16 + v + m_hi;
        const float val = (acc[mt][nt][v] + bcol) * out_scale;
        if (out_bf) {
          if (transpose_out) out_bf[(size_t)col * S_LEN + row] = (__bf16)val; // V^T [E][S]
          else               out_bf[(size_t)row * EMB + col]   = (__bf16)val; // [S][E]
        }
        if (out_f32) out_f32[(size_t)row * EMB + col] = val;
      }
    }
  }
}

// ---------------------------------------------------------------------------
// Sparse attention: one workgroup per (64-row block, head).
// Phase 1: 64x64 diagonal score block via WMMA (Q pre-scaled by 1/sqrt(D)).
// Phase 2: per-row softmax; scan mask row for rare off-block entries,
//          compute their scores scalarly; write fp32 attn rows.
// Phase 3: ctx = P @ V via WMMA (+ scalar extras), store ctx as bf16.
// ---------------------------------------------------------------------------
__global__ __launch_bounds__(128) void sparse_attn_kernel(
    const __bf16* __restrict__ Qb, const __bf16* __restrict__ Kb,
    const __bf16* __restrict__ Vt, const unsigned char* __restrict__ mask,
    float* __restrict__ attn_out, __bf16* __restrict__ ctx)
{
  __shared__ float  sc[BLK][BLK];
  __shared__ __bf16 pb[BLK][BLK];
  __shared__ int    extk[BLK][MAXE];
  __shared__ float  extp[BLK][MAXE];
  __shared__ int    extn[BLK];

  const int lane = threadIdx.x & 31;
  const int wave = threadIdx.x >> 5;
  const int h    = blockIdx.x & (HEADS - 1);
  const int blk  = blockIdx.x >> 4;
  const int q0   = blk * BLK;
  const int row0 = q0 + wave * 16;
  const int dbase = h * HD;

  // ---- Phase 1: scores for the diagonal block (K-dim = D = 64 -> 2 chunks)
  {
    v16bf a0 = tile_load_bf16(Qb, EMB, row0, dbase, lane);
    v16bf a1 = tile_load_bf16(Qb, EMB, row0, dbase + 32, lane);
    const int n_lane = lane & 15;
    const int m_hi = (lane >> 4) << 3;
#pragma unroll
    for (int nt = 0; nt < 4; ++nt) {
      v16bf b0 = tile_load_bf16(Kb, EMB, q0 + nt * 16, dbase, lane);
      v16bf b1 = tile_load_bf16(Kb, EMB, q0 + nt * 16, dbase + 32, lane);
      v8f c = {};
      c = wmma_bf16(a0, b0, c);
      c = wmma_bf16(a1, b1, c);
#pragma unroll
      for (int v = 0; v < 8; ++v)
        sc[wave * 16 + v + m_hi][nt * 16 + n_lane] = c[v];
    }
  }
  __syncthreads();

  // ---- Phase 2: softmax per row (64 row-owner threads)
  if (threadIdx.x < BLK) {
    const int r = threadIdx.x;
    const int rg = q0 + r;
    const unsigned char* mrow = mask + (size_t)rg * S_LEN;
    int ne = 0;
    int ek[MAXE];
    for (int c16 = 0; c16 < S_LEN / 16; ++c16) {
      if ((c16 >> 2) == blk) continue;            // own diagonal block
      const uint4 m16 = *(const uint4*)(mrow + c16 * 16);
      if (m16.x | m16.y | m16.z | m16.w) {
        for (int b = 0; b < 16; ++b)
          if (mrow[c16 * 16 + b] && ne < MAXE) ek[ne++] = c16 * 16 + b;
      }
    }
    float es[MAXE];
    const __bf16* qrow = Qb + (size_t)rg * EMB + dbase;    // Q already * 1/8
    for (int e = 0; e < ne; ++e) {
      const __bf16* krow = Kb + (size_t)ek[e] * EMB + dbase;
      float s = 0.f;
      for (int d = 0; d < HD; ++d) s += (float)qrow[d] * (float)krow[d];
      es[e] = s;
    }
    float mx = -1e30f;
    for (int k = 0; k < BLK; ++k) mx = fmaxf(mx, sc[r][k]);
    for (int e = 0; e < ne; ++e) mx = fmaxf(mx, es[e]);
    float sum = 0.f;
    for (int k = 0; k < BLK; ++k) sum += __expf(sc[r][k] - mx);
    for (int e = 0; e < ne; ++e) sum += __expf(es[e] - mx);
    const float inv = 1.f / sum;
    float* arow = attn_out + ((size_t)h * S_LEN + rg) * S_LEN;
    for (int k = 0; k < BLK; ++k) {
      const float p = __expf(sc[r][k] - mx) * inv;
      arow[q0 + k] = p;
      pb[r][k] = (__bf16)p;
    }
    for (int e = 0; e < ne; ++e) {
      const float p = __expf(es[e] - mx) * inv;
      arow[ek[e]] = p;
      extk[r][e] = ek[e];
      extp[r][e] = p;
    }
    extn[r] = ne;
  }
  __syncthreads();

  // ---- Phase 3: context = P @ V (block) + extras; store bf16 ctx [S][E]
  {
    v16bf p0 = tile_load_bf16(&pb[0][0], BLK, wave * 16, 0, lane);
    v16bf p1 = tile_load_bf16(&pb[0][0], BLK, wave * 16, 32, lane);
    const int n_lane = lane & 15;
    const int m_hi = (lane >> 4) << 3;
#pragma unroll
    for (int nt = 0; nt < 4; ++nt) {
      // B tile (k x d): column d of V is row d of V^T -> contiguous in s_k.
      v16bf b0 = tile_load_bf16(Vt, S_LEN, dbase + nt * 16, q0, lane);
      v16bf b1 = tile_load_bf16(Vt, S_LEN, dbase + nt * 16, q0 + 32, lane);
      v8f c = {};
      c = wmma_bf16(p0, b0, c);
      c = wmma_bf16(p1, b1, c);
      const int drow = dbase + nt * 16 + n_lane;
#pragma unroll
      for (int v = 0; v < 8; ++v) {
        const int r = wave * 16 + v + m_hi;
        float add = 0.f;
        const int ne = extn[r];
        for (int e = 0; e < ne; ++e)
          add += extp[r][e] * (float)Vt[(size_t)drow * S_LEN + extk[r][e]];
        ctx[(size_t)(q0 + r) * EMB + drow] = (__bf16)(c[v] + add);
      }
    }
  }
}

// ---------------------------------------------------------------------------
extern "C" void kernel_launch(void* const* d_in, const int* in_sizes, int n_in,
                              void* d_out, int out_size, void* d_ws, size_t ws_size,
                              hipStream_t stream)
{
  const float* x  = (const float*)d_in[0];
  const float* Wq = (const float*)d_in[1];
  const float* bq = (const float*)d_in[2];
  const float* Wk = (const float*)d_in[3];
  const float* bk = (const float*)d_in[4];
  const float* Wv = (const float*)d_in[5];
  const float* bv = (const float*)d_in[6];
  const float* Wo = (const float*)d_in[7];
  const float* bo = (const float*)d_in[8];
  const unsigned char* mask = (const unsigned char*)d_in[9];

  float* out  = (float*)d_out;
  float* attn = out + (size_t)S_LEN * EMB;   // outputs concatenated: out, attn

  const size_t NE = (size_t)S_LEN * EMB;     // 4M elements
  const size_t NW = (size_t)EMB * EMB;       // 1M elements
  __bf16* ws   = (__bf16*)d_ws;
  __bf16* xb   = ws;                          // [S][E] bf16
  __bf16* Wqb  = ws + NE;                     // [E][E] bf16 (x4)
  __bf16* Wkb  = Wqb + NW;
  __bf16* Wvb  = Wkb + NW;
  __bf16* Wob  = Wvb + NW;
  __bf16* Qb   = Wob + NW;                    // [S][E] bf16, * 1/sqrt(D)
  __bf16* Kb   = Qb + NE;                     // [S][E] bf16
  __bf16* Vt   = Kb + NE;                     // [E][S] bf16 (transposed)
  __bf16* ctx  = Vt + NE;                     // [S][E] bf16

  // One-time fp32 -> bf16 conversions (all L2-resident afterwards).
  cvt_bf16_kernel<<<1024, 256, 0, stream>>>(x,  xb,  (long)NE / 8);
  cvt_bf16_kernel<<<256, 256, 0, stream>>>(Wq, Wqb, (long)NW / 8);
  cvt_bf16_kernel<<<256, 256, 0, stream>>>(Wk, Wkb, (long)NW / 8);
  cvt_bf16_kernel<<<256, 256, 0, stream>>>(Wv, Wvb, (long)NW / 8);
  cvt_bf16_kernel<<<256, 256, 0, stream>>>(Wo, Wob, (long)NW / 8);

  // attn is exactly 0 outside the sparse pattern -> bulk zero (NT stores).
  zero_f32_kernel<<<4096, 256, 0, stream>>>(
      (v4f*)attn, (long)HEADS * S_LEN * S_LEN / 4);

  const dim3 gemm_grid(512), attn_grid(1024), blkdim(128);
  proj_gemm_kernel<<<gemm_grid, blkdim, 0, stream>>>(xb, Wqb, bq, Qb, nullptr, 0.125f, 0);
  proj_gemm_kernel<<<gemm_grid, blkdim, 0, stream>>>(xb, Wkb, bk, Kb, nullptr, 1.0f, 0);
  proj_gemm_kernel<<<gemm_grid, blkdim, 0, stream>>>(xb, Wvb, bv, Vt, nullptr, 1.0f, 1);
  sparse_attn_kernel<<<attn_grid, blkdim, 0, stream>>>(Qb, Kb, Vt, mask, attn, ctx);
  proj_gemm_kernel<<<gemm_grid, blkdim, 0, stream>>>(ctx, Wob, bo, nullptr, out, 1.0f, 0);
}
